// BlockwiseAttention_34179349741875
// MI455X (gfx1250) — matrix operand order repaired and checked by reading, not compiled, not run on hardware
//
#include <hip/hip_runtime.h>
#include <hip/hip_bf16.h>

typedef __attribute__((ext_vector_type(16))) _Float16 v16h;
typedef __attribute__((ext_vector_type(8)))  _Float16 v8h;
typedef __attribute__((ext_vector_type(8)))  float    v8f;

#define S_LEN 2048
#define EMB_  1024
#define HEADS_ 16
#define HDIM  64

static __device__ __forceinline__ v16h cat16(v8h a, v8h b) {
  return __builtin_shufflevector(a, b, 0,1,2,3,4,5,6,7,8,9,10,11,12,13,14,15);
}
static __device__ __forceinline__ v8f wmma16(v16h a, v16h b, v8f c) {
  return __builtin_amdgcn_wmma_f32_16x16x32_f16(false, a, false, b, (short)0, c, false, false);
}
// One-lane 16B async copy global -> LDS (ASYNCcnt-tracked, no VGPR data path)
static __device__ __forceinline__ void async_copy16(const _Float16* gp, _Float16* lp) {
  unsigned lo = (unsigned)(uintptr_t)lp;   // low 32 bits of generic ptr = LDS byte addr
  asm volatile("global_load_async_to_lds_b128 %0, %1, off"
               :: "v"(lo), "v"(gp) : "memory");
}
static __device__ __forceinline__ void async_wait0() {
  asm volatile("s_wait_asynccnt 0" ::: "memory");
}

// ---------------------------------------------------------------------------
// Kernel 1: qkv = x @ w_qkv (fp32 in, f16 out to [t][b][h][s][d]).
// 128x128 tile, 8 waves (2x4), K step 64 (16 WMMAs per barrier pair).
// ---------------------------------------------------------------------------
__global__ __launch_bounds__(256) void qkv_gemm_kernel(
    const float* __restrict__ X, const float* __restrict__ W,
    _Float16* __restrict__ qkv)
{
  __shared__ _Float16 lA[128 * 72];   // [m][k], stride 72 halves (144B)
  __shared__ _Float16 lB[128 * 72];   // [n][k] transposed
  const int tid  = threadIdx.x;
  const int lane = tid & 31, wave = tid >> 5;
  const int half = lane >> 4, l16 = lane & 15;
  const int wm = wave >> 2, wn = wave & 3;
  const int bm = blockIdx.y * 128, bn = blockIdx.x * 128;

  v8f zero = {};
  v8f acc[4][2];
  for (int mt = 0; mt < 4; ++mt)
    for (int nt = 0; nt < 2; ++nt) acc[mt][nt] = zero;

  for (int k0 = 0; k0 < EMB_; k0 += 64) {
    // A tile 128x64 f32 -> f16 LDS (conversion forces manual path)
    for (int i = 0; i < 8; ++i) {
      int idx = tid + i * 256;          // float4 index 0..2047
      int r = idx >> 4, c4 = idx & 15;  // 16 float4 per row
      float4 f = ((const float4*)(X + (size_t)(bm + r) * EMB_ + k0))[c4];
      _Float16* d = &lA[r * 72 + c4 * 4];
      d[0] = (_Float16)f.x; d[1] = (_Float16)f.y;
      d[2] = (_Float16)f.z; d[3] = (_Float16)f.w;
    }
    // B tile 64x128 f32 -> transposed f16 LDS [n][k]
    for (int i = 0; i < 8; ++i) {
      int idx = tid + i * 256;
      int r = idx >> 5, c4 = idx & 31;  // 32 float4 per row
      float4 f = ((const float4*)(W + (size_t)(k0 + r) * 3072 + bn))[c4];
      int n = c4 * 4;
      lB[(n + 0) * 72 + r] = (_Float16)f.x;
      lB[(n + 1) * 72 + r] = (_Float16)f.y;
      lB[(n + 2) * 72 + r] = (_Float16)f.z;
      lB[(n + 3) * 72 + r] = (_Float16)f.w;
    }
    __syncthreads();

    for (int kc = 0; kc < 2; ++kc) {
      v16h bf[2];
      for (int nt = 0; nt < 2; ++nt) {
        int n = wn * 32 + nt * 16 + l16;
        const v8h* p = (const v8h*)&lB[n * 72 + kc * 32 + half * 16];
        bf[nt] = cat16(p[0], p[1]);
      }
      for (int mt = 0; mt < 4; ++mt) {
        int m = wm * 64 + mt * 16 + l16;
        v16h af = cat16(*(const v8h*)&lA[m * 72 + kc * 32 + half * 16],
                        *(const v8h*)&lA[m * 72 + kc * 32 + half * 16 + 8]);
        for (int nt = 0; nt < 2; ++nt)
          acc[mt][nt] = wmma16(af, bf[nt], acc[mt][nt]);
      }
    }
    __syncthreads();
  }

  for (int mt = 0; mt < 4; ++mt)
    for (int nt = 0; nt < 2; ++nt) {
      int col = bn + wn * 32 + nt * 16 + l16;
      int t = col >> 10, hh = (col >> 6) & 15, d = col & 63;
      for (int r = 0; r < 8; ++r) {
        int M = bm + wm * 64 + mt * 16 + r + 8 * half;
        int b = M >> 11, s = M & (S_LEN - 1);
        qkv[((((size_t)t * 2 + b) * HEADS_ + hh) * S_LEN + s) * HDIM + d] =
            (_Float16)acc[mt][nt][r];
      }
    }
}

// ---------------------------------------------------------------------------
// Kernel 2: in-place RoPE on Q and K halves of qkv buffer.
// ---------------------------------------------------------------------------
__global__ __launch_bounds__(256) void rope_kernel(_Float16* qkv)
{
  int idx = blockIdx.x * 256 + threadIdx.x;
  if (idx >= 2 * HEADS_ * S_LEN * 32) return;
  int i2 = idx & 31;
  int s  = (idx >> 5) & (S_LEN - 1);
  int bh = idx >> 16;
  size_t base = ((size_t)bh * S_LEN + s) * HDIM;
  int d0 = i2 * 2, d1 = d0 + 1;
  const float LN10K = 9.210340371976184f;
  float f0 = __expf(-(float)(d0 & 31) * (LN10K / 32.0f));
  float f1 = __expf(-(float)(d1 & 31) * (LN10K / 32.0f));
  float p = (float)s;
  float s0, c0, s1, c1;
  __sincosf(p * f0, &s0, &c0);
  __sincosf(p * f1, &s1, &c1);
  _Float16* Q = qkv + base;
  _Float16* K = qkv + (size_t)2 * HEADS_ * S_LEN * HDIM + base;
  float q0 = (float)Q[d0], q1 = (float)Q[d1];
  Q[d0] = (_Float16)(q0 * c0 - q1 * s0);
  Q[d1] = (_Float16)(q1 * c1 + q0 * s1);
  float k0 = (float)K[d0], k1 = (float)K[d1];
  K[d0] = (_Float16)(k0 * c0 - k1 * s0);
  K[d1] = (_Float16)(k1 * c1 + k0 * s1);
}

// ---------------------------------------------------------------------------
// Kernel 3: flash-style masked attention (reference's inverted mask).
// K tile staged via async global->LDS; V transposed manually (needs VGPRs).
// ---------------------------------------------------------------------------
__global__ __launch_bounds__(256) void attn_kernel(
    const _Float16* __restrict__ qkv, _Float16* __restrict__ aout)
{
  __shared__ _Float16 lK[128 * 72];      // [t][d]
  __shared__ _Float16 lV[64 * 136];      // [d][t] transposed
  __shared__ _Float16 lP[8 * 16 * 136];  // per-wave P tile
  const int tid  = threadIdx.x;
  const int lane = tid & 31, wave = tid >> 5;
  const int half = lane >> 4, l16 = lane & 15;
  const int bh = blockIdx.y;
  const int q0 = blockIdx.x * 128;
  const _Float16* Qg = qkv + (size_t)bh * S_LEN * HDIM;
  const _Float16* Kg = qkv + (size_t)(32 + bh) * S_LEN * HDIM;
  const _Float16* Vg = qkv + (size_t)(64 + bh) * S_LEN * HDIM;
  _Float16* P = &lP[wave * 16 * 136];

  const int qrow = q0 + wave * 16 + l16;
  v16h qf[2];
  for (int c = 0; c < 2; ++c) {
    const v8h* p0 = (const v8h*)(Qg + (size_t)qrow * HDIM + c * 32 + half * 16);
    const v8h* p1 = (const v8h*)(Qg + (size_t)qrow * HDIM + c * 32 + half * 16 + 8);
    qf[c] = cat16(*p0, *p1);
  }

  v8f zero = {};
  v8f O[4];
  for (int dt = 0; dt < 4; ++dt) O[dt] = zero;
  float mrow[8], lrow[8];
  for (int r = 0; r < 8; ++r) { mrow[r] = -3e30f; lrow[r] = 0.0f; }

  for (int tb = 0; tb < S_LEN; tb += 128) {
    // K tile -> LDS via async copy (ASYNCcnt path)
    for (int i = 0; i < 4; ++i) {
      int ch = tid + i * 256;
      int r = ch >> 3, cc = ch & 7;
      async_copy16(Kg + (size_t)(tb + r) * HDIM + cc * 8, &lK[r * 72 + cc * 8]);
    }
    // V tile -> LDS transposed [d][t] (manual: needs per-element shuffle)
    for (int i = 0; i < 4; ++i) {
      int ch = tid + i * 256;
      int r = ch >> 3, cc = ch & 7;
      v8h v = *(const v8h*)(Vg + (size_t)(tb + r) * HDIM + cc * 8);
      for (int j = 0; j < 8; ++j) lV[(cc * 8 + j) * 136 + r] = v[j];
    }
    async_wait0();
    __syncthreads();

    // scores = Q @ K^T
    v8f Sc[8];
    for (int nt = 0; nt < 8; ++nt) {
      int n = nt * 16 + l16;
      v16h b0 = cat16(*(const v8h*)&lK[n * 72 + half * 16],
                      *(const v8h*)&lK[n * 72 + half * 16 + 8]);
      v16h b1 = cat16(*(const v8h*)&lK[n * 72 + 32 + half * 16],
                      *(const v8h*)&lK[n * 72 + 32 + half * 16 + 8]);
      v8f s = zero;
      s = wmma16(qf[0], b0, s);
      s = wmma16(qf[1], b1, s);
      Sc[nt] = s;
    }

    // mask + online softmax (row = r + 8*half, col = tb + nt*16 + l16)
    for (int r = 0; r < 8; ++r) {
      int i = q0 + wave * 16 + r + 8 * half;
      float mx = -3e30f;
      for (int nt = 0; nt < 8; ++nt) {
        int j = tb + nt * 16 + l16;
        float sv = Sc[nt][r] * 0.125f;
        int dd = i - j; if (dd < 0) dd = -dd;
        bool allowed = (((i >> 8) == (j >> 8)) || (i < 4) || (j < 4) ||
                        (dd <= 128)) && (j <= i);
        if (allowed) sv = -1e30f;          // reference masks the ALLOWED set
        Sc[nt][r] = sv;
        mx = fmaxf(mx, sv);
      }
      for (int off = 1; off < 16; off <<= 1)
        mx = fmaxf(mx, __shfl_xor(mx, off, 32));
      float newm = fmaxf(mrow[r], mx);
      float corr = __expf(mrow[r] - newm);
      float rs = 0.0f;
      for (int nt = 0; nt < 8; ++nt) {
        float sv = Sc[nt][r];
        float pv = (sv < -5e29f) ? 0.0f : __expf(sv - newm);
        Sc[nt][r] = pv;
        rs += pv;
      }
      for (int off = 1; off < 16; off <<= 1)
        rs += __shfl_xor(rs, off, 32);
      lrow[r] = lrow[r] * corr + rs;
      mrow[r] = newm;
      for (int dt = 0; dt < 4; ++dt) O[dt][r] *= corr;
    }

    // P (C layout) -> per-wave LDS -> A fragments
    for (int nt = 0; nt < 8; ++nt)
      for (int r = 0; r < 8; ++r)
        P[(r + 8 * half) * 136 + nt * 16 + l16] = (_Float16)Sc[nt][r];
    asm volatile("s_wait_dscnt 0" ::: "memory");

    // O += P @ V
    for (int kc = 0; kc < 4; ++kc) {
      v16h af = cat16(*(const v8h*)&P[l16 * 136 + kc * 32 + half * 16],
                      *(const v8h*)&P[l16 * 136 + kc * 32 + half * 16 + 8]);
      for (int dt = 0; dt < 4; ++dt) {
        v16h bf = cat16(
            *(const v8h*)&lV[(dt * 16 + l16) * 136 + kc * 32 + half * 16],
            *(const v8h*)&lV[(dt * 16 + l16) * 136 + kc * 32 + half * 16 + 8]);
        O[dt] = wmma16(af, bf, O[dt]);
      }
    }
    __syncthreads();
  }

  const int b = bh >> 4, hh = bh & 15;
  for (int dt = 0; dt < 4; ++dt)
    for (int r = 0; r < 8; ++r) {
      int i = q0 + wave * 16 + r + 8 * half;
      float o = O[dt][r] / lrow[r];
      aout[((size_t)b * S_LEN + i) * EMB_ + hh * HDIM + dt * 16 + l16] =
          (_Float16)o;
    }
}

// ---------------------------------------------------------------------------
// Kernel 4: out = attn_out @ w_out  (f16 A via async LDS copy, f32 W -> fp32)
// K step 64.
// ---------------------------------------------------------------------------
__global__ __launch_bounds__(256) void out_gemm_kernel(
    const _Float16* __restrict__ A, const float* __restrict__ W,
    float* __restrict__ out)
{
  __shared__ _Float16 lA[128 * 72];
  __shared__ _Float16 lB[128 * 72];
  const int tid  = threadIdx.x;
  const int lane = tid & 31, wave = tid >> 5;
  const int half = lane >> 4, l16 = lane & 15;
  const int wm = wave >> 2, wn = wave & 3;
  const int bm = blockIdx.y * 128, bn = blockIdx.x * 128;

  v8f zero = {};
  v8f acc[4][2];
  for (int mt = 0; mt < 4; ++mt)
    for (int nt = 0; nt < 2; ++nt) acc[mt][nt] = zero;

  for (int k0 = 0; k0 < EMB_; k0 += 64) {
    // A tile 128x64 f16: pure copy -> async global->LDS
    for (int i = 0; i < 4; ++i) {
      int ch = tid + i * 256;          // 1024 chunks of 8 halves
      int r = ch >> 3, cc = ch & 7;    // 8 chunks per row
      async_copy16(A + (size_t)(bm + r) * EMB_ + k0 + cc * 8,
                   &lA[r * 72 + cc * 8]);
    }
    // B tile 64x128 f32 -> transposed f16 LDS [n][k]
    for (int i = 0; i < 8; ++i) {
      int idx = tid + i * 256;
      int r = idx >> 5, c4 = idx & 31;
      float4 f = ((const float4*)(W + (size_t)(k0 + r) * EMB_ + bn))[c4];
      int n = c4 * 4;
      lB[(n + 0) * 72 + r] = (_Float16)f.x;
      lB[(n + 1) * 72 + r] = (_Float16)f.y;
      lB[(n + 2) * 72 + r] = (_Float16)f.z;
      lB[(n + 3) * 72 + r] = (_Float16)f.w;
    }
    async_wait0();
    __syncthreads();

    for (int kc = 0; kc < 2; ++kc) {
      v16h bf[2];
      for (int nt = 0; nt < 2; ++nt) {
        int n = wn * 32 + nt * 16 + l16;
        const v8h* p = (const v8h*)&lB[n * 72 + kc * 32 + half * 16];
        bf[nt] = cat16(p[0], p[1]);
      }
      for (int mt = 0; mt < 4; ++mt) {
        int m = wm * 64 + mt * 16 + l16;
        v16h af = cat16(*(const v8h*)&lA[m * 72 + kc * 32 + half * 16],
                        *(const v8h*)&lA[m * 72 + kc * 32 + half * 16 + 8]);
        for (int nt = 0; nt < 2; ++nt)
          acc[mt][nt] = wmma16(af, bf[nt], acc[mt][nt]);
      }
    }
    __syncthreads();
  }

  for (int mt = 0; mt < 4; ++mt)
    for (int nt = 0; nt < 2; ++nt) {
      int col = bn + wn * 32 + nt * 16 + l16;
      for (int r = 0; r < 8; ++r) {
        int M = bm + wm * 64 + mt * 16 + r + 8 * half;
        out[(size_t)M * EMB_ + col] = acc[mt][nt][r];
      }
    }
}

// ---------------------------------------------------------------------------
extern "C" void kernel_launch(void* const* d_in, const int* in_sizes, int n_in,
                              void* d_out, int out_size, void* d_ws, size_t ws_size,
                              hipStream_t stream) {
  (void)in_sizes; (void)n_in; (void)out_size; (void)ws_size;
  const float* x     = (const float*)d_in[0];
  const float* w_qkv = (const float*)d_in[1];
  const float* w_out = (const float*)d_in[2];
  float* out = (float*)d_out;

  _Float16* qkv  = (_Float16*)d_ws;                              // 24 MB
  _Float16* aout = (_Float16*)((char*)d_ws +
                   (size_t)3 * 2 * HEADS_ * S_LEN * HDIM * sizeof(_Float16)); // 8 MB

  qkv_gemm_kernel<<<dim3(24, 32), 256, 0, stream>>>(x, w_qkv, qkv);
  rope_kernel<<<(2 * HEADS_ * S_LEN * 32 + 255) / 256, 256, 0, stream>>>(qkv);
  attn_kernel<<<dim3(16, 32), 256, 0, stream>>>(qkv, aout);
  out_gemm_kernel<<<dim3(8, 32), 256, 0, stream>>>(aout, w_out, out);
}
